// Encoder_10488310136844
// MI455X (gfx1250) — matrix operand order repaired and checked by reading, not compiled, not run on hardware
//
#include <hip/hip_runtime.h>

typedef unsigned short u16;
typedef unsigned int   u32;
typedef unsigned long long u64;
typedef __attribute__((ext_vector_type(16))) __bf16 v16bf;
typedef __attribute__((ext_vector_type(8)))  float  v8f;
typedef __attribute__((ext_vector_type(8)))  u16    u16x8;

#define B_    4
#define S_    1024
#define E_    512
#define H_    8
#define L_    4
#define DH_   64
#define OBS_  128
#define NROWS (B_ * S_)
#define EPS_  1e-6f

// ---------------- helpers ----------------
__device__ __forceinline__ u16 f2bf(float x) {
  u32 u = __float_as_uint(x);
  u32 r = u + 0x7fffu + ((u >> 16) & 1u);
  return (u16)(r >> 16);
}
__device__ __forceinline__ float bf2f(u16 u) { return __uint_as_float(((u32)u) << 16); }
__device__ __forceinline__ float gelu_f(float x) {
  float x3 = x * x * x;
  return 0.5f * x * (1.f + tanhf(0.7978845608028654f * (x + 0.044715f * x3)));
}
__device__ __forceinline__ float silu_f(float x) { return x / (1.f + expf(-x)); }
__device__ __forceinline__ float head_logk(int h) {
  float t0 = logf(1.f / 32.f), t1 = logf(1.f / 512.f);
  float lin = t0 + (t1 - t0) * ((float)h / (float)(H_ - 1));
  return logf(1.f - expf(lin));
}
__device__ __forceinline__ v8f v8f_zero() {
  v8f z = {0.f, 0.f, 0.f, 0.f, 0.f, 0.f, 0.f, 0.f};
  return z;
}
union FragB { v16bf v; u16x8 h[2]; };

// async copy of 32B (two b128, INST_OFFSET applies to both LDS and global side)
__device__ __forceinline__ void async_cp32(unsigned lds_addr, const void* gptr) {
  u64 ga = (u64)gptr;
  asm volatile(
      "global_load_async_to_lds_b128 %0, %1, off\n\t"
      "global_load_async_to_lds_b128 %0, %1, off offset:16"
      :: "v"(lds_addr), "v"(ga) : "memory");
}
// async copy of 16B
__device__ __forceinline__ void async_cp16(unsigned lds_addr, const void* gptr) {
  u64 ga = (u64)gptr;
  asm volatile("global_load_async_to_lds_b128 %0, %1, off"
               :: "v"(lds_addr), "v"(ga) : "memory");
}
__device__ __forceinline__ unsigned lds_off(const void* p) {
  return (unsigned)(u64)p;  // flat shared addr: low 32 bits = LDS offset
}

// ---------------- tiny kernels ----------------
__global__ void seg_kernel(const int* __restrict__ dones, int* __restrict__ seg) {
  int b = threadIdx.x;
  if (b >= B_) return;
  int acc = 0;
  for (int s = 0; s < S_; ++s) { acc += dones[b * S_ + s]; seg[b * S_ + s] = acc; }
}

// in: [K][N] f32 row-major -> out: [N][K] bf16 row-major (pre-transposed weights)
__global__ void wconv_kernel(const float* __restrict__ in, u16* __restrict__ out, int K, int N) {
  int total = K * N;
  for (int i = blockIdx.x * blockDim.x + threadIdx.x; i < total; i += gridDim.x * blockDim.x) {
    int k = i / N, n = i % N;
    out[(size_t)n * K + k] = f2bf(in[i]);
  }
}

// rmsnorm(a (+ b), scale) over K columns; optional f32 and bf16 outputs
__global__ __launch_bounds__(256) void rmsnorm_kernel(
    const float* __restrict__ a, const float* __restrict__ b,
    const float* __restrict__ scale, int K, float* outF, u16* outB) {
  __shared__ float red[256];
  int row = blockIdx.x, t = threadIdx.x;
  const float* ap = a + (size_t)row * K;
  const float* bp = b ? b + (size_t)row * K : nullptr;
  float ss = 0.f;
  for (int e = t; e < K; e += 256) {
    float v = ap[e] + (bp ? bp[e] : 0.f);
    ss += v * v;
  }
  red[t] = ss; __syncthreads();
  for (int off = 128; off > 0; off >>= 1) { if (t < off) red[t] += red[t + off]; __syncthreads(); }
  float r = rsqrtf(red[0] / (float)K + EPS_);
  for (int e = t; e < K; e += 256) {
    float v = (ap[e] + (bp ? bp[e] : 0.f)) * r * scale[e];
    if (outF) outF[(size_t)row * K + e] = v;
    if (outB) outB[(size_t)row * K + e] = f2bf(v);
  }
}

__global__ void silu_mul_kernel(const float* __restrict__ a, const float* __restrict__ b,
                                u16* __restrict__ o, int n) {
  for (int i = blockIdx.x * blockDim.x + threadIdx.x; i < n; i += gridDim.x * blockDim.x)
    o[i] = f2bf(silu_f(a[i]) * b[i]);
}

// per-row per-head (64-wide) groupnorm of ret, then silu(g) * y -> bf16
__global__ __launch_bounds__(256) void gn_silu_kernel(
    const float* __restrict__ retv, const float* __restrict__ gf,
    const float* __restrict__ gns, const float* __restrict__ gnb, u16* __restrict__ outB) {
  __shared__ float sv[E_];
  __shared__ float mug[H_], rsg[H_];
  int row = blockIdx.x, t = threadIdx.x;
  for (int e = t; e < E_; e += 256) sv[e] = retv[(size_t)row * E_ + e];
  __syncthreads();
  if (t < H_) {
    float s = 0.f, sq = 0.f;
    for (int d = 0; d < DH_; ++d) { float v = sv[t * DH_ + d]; s += v; sq += v * v; }
    float mu = s / (float)DH_;
    mug[t] = mu;
    rsg[t] = rsqrtf(sq / (float)DH_ - mu * mu + EPS_);
  }
  __syncthreads();
  for (int e = t; e < E_; e += 256) {
    int g = e >> 6;
    float y = (sv[e] - mug[g]) * rsg[g] * gns[e] + gnb[e];
    outB[(size_t)row * E_ + e] = f2bf(silu_f(gf[(size_t)row * E_ + e]) * y);
  }
}

// v_loc[row] = rmsnorm(z, head_ln) . head_w2 + head_b2
__global__ __launch_bounds__(256) void value_head_kernel(
    const float* __restrict__ z, const float* __restrict__ hln,
    const float* __restrict__ w2, const float* __restrict__ b2, float* __restrict__ vloc) {
  __shared__ float red[256];
  int row = blockIdx.x, t = threadIdx.x;
  const float* zp = z + (size_t)row * E_;
  float ss = 0.f;
  for (int e = t; e < E_; e += 256) { float v = zp[e]; ss += v * v; }
  red[t] = ss; __syncthreads();
  for (int off = 128; off > 0; off >>= 1) { if (t < off) red[t] += red[t + off]; __syncthreads(); }
  float r = rsqrtf(red[0] / (float)E_ + EPS_);
  __syncthreads();
  float dot = 0.f;
  for (int e = t; e < E_; e += 256) dot += zp[e] * r * hln[e] * w2[e];
  red[t] = dot; __syncthreads();
  for (int off = 128; off > 0; off >>= 1) { if (t < off) red[t] += red[t + off]; __syncthreads(); }
  if (t == 0) vloc[row] = red[0] + b2[0];
}

// ---------------- bf16 WMMA GEMM with async-to-LDS double buffering ----------------
// C[M,N] = act(A[M,K] @ Bt[N,K]^T + bias) * scale ; A,Bt bf16 row-major
// 256 threads = 8 waves; wave -> (wm = w&3 -> 32 rows, wn = w>>2 -> 64 cols)
__global__ __launch_bounds__(256) void gemm_bf16_kernel(
    const u16* __restrict__ A, const u16* __restrict__ Bt, int M, int N, int K,
    float* outF, u16* outB, const float* __restrict__ bias, int act, float scale) {
  __shared__ u16 As[2][128][32];
  __shared__ u16 Bs[2][128][32];
  int t = threadIdx.x;
  int w = t >> 5, l = t & 31;
  int wm = w & 3, wn = w >> 2;
  int khalf = l >> 4, lr = l & 15;
  int blockRow = blockIdx.y * 128, blockCol = blockIdx.x * 128;
  int lrow = t >> 1, lcol = (t & 1) * 16;

  const u16* agBase = A + (size_t)(blockRow + lrow) * K + lcol;
  const u16* bgBase = Bt + (size_t)(blockCol + lrow) * K + lcol;
  unsigned ldsA[2] = { lds_off(&As[0][lrow][lcol]), lds_off(&As[1][lrow][lcol]) };
  unsigned ldsB[2] = { lds_off(&Bs[0][lrow][lcol]), lds_off(&Bs[1][lrow][lcol]) };

  v8f acc[2][4];
#pragma unroll
  for (int i = 0; i < 2; ++i)
#pragma unroll
    for (int j = 0; j < 4; ++j) acc[i][j] = v8f_zero();

  int nk = K >> 5;
  // prologue: tile 0 -> buffer 0 (4 async b128 per thread)
  async_cp32(ldsA[0], agBase);
  async_cp32(ldsB[0], bgBase);

  for (int ki = 0; ki < nk; ++ki) {
    int buf = ki & 1;
    __syncthreads();  // all waves done reading buf^1 -> safe to refill it
    if (ki + 1 < nk) {
      async_cp32(ldsA[buf ^ 1], agBase + ((ki + 1) << 5));
      async_cp32(ldsB[buf ^ 1], bgBase + ((ki + 1) << 5));
      asm volatile("s_wait_asynccnt 0x4" ::: "memory");  // tile ki landed (in-order)
    } else {
      asm volatile("s_wait_asynccnt 0x0" ::: "memory");
    }
    __syncthreads();  // tile ki visible to all waves

    FragB a[2], bb[4];
#pragma unroll
    for (int mt = 0; mt < 2; ++mt) {
      int r = wm * 32 + mt * 16 + lr;
      a[mt].h[0] = *(const u16x8*)&As[buf][r][khalf * 8];
      a[mt].h[1] = *(const u16x8*)&As[buf][r][16 + khalf * 8];
    }
#pragma unroll
    for (int nt = 0; nt < 4; ++nt) {
      int c = wn * 64 + nt * 16 + lr;
      bb[nt].h[0] = *(const u16x8*)&Bs[buf][c][khalf * 8];
      bb[nt].h[1] = *(const u16x8*)&Bs[buf][c][16 + khalf * 8];
    }
#pragma unroll
    for (int mt = 0; mt < 2; ++mt)
#pragma unroll
      for (int nt = 0; nt < 4; ++nt)
        acc[mt][nt] = __builtin_amdgcn_wmma_f32_16x16x32_bf16(
            false, a[mt].v, false, bb[nt].v, (short)0, acc[mt][nt], false, false);
  }

#pragma unroll
  for (int mt = 0; mt < 2; ++mt)
#pragma unroll
    for (int nt = 0; nt < 4; ++nt)
#pragma unroll
      for (int j = 0; j < 8; ++j) {
        int r = blockRow + wm * 32 + mt * 16 + j + khalf * 8;
        int c = blockCol + wn * 64 + nt * 16 + lr;
        float v = acc[mt][nt][j];
        if (bias) v += bias[c];
        if (act == 1) v = gelu_f(v);
        v *= scale;
        if (outF) outF[(size_t)r * N + c] = v;
        if (outB) outB[(size_t)r * N + c] = f2bf(v);
      }
}

// ---------------- retention (flash-style, per (b,h), 64-query block) ----------------
__global__ __launch_bounds__(128) void retention_kernel(
    const u16* __restrict__ qb, const u16* __restrict__ kb, const u16* __restrict__ vb,
    const int* __restrict__ ts, const int* __restrict__ seg,
    const float* __restrict__ hstate, int layer, float* __restrict__ ret) {
  __shared__ u16 Qs[64][64];   // q (A layout: rows=n, cols=d)
  __shared__ u16 Ks[64][64];   // k (B layout: rows=m, cols=d)
  __shared__ u16 Vt[64][64];   // v^T (B layout: rows=e, cols=m)
  __shared__ u16 Ps[64][64];   // masked scores (A layout rows=n, cols=m); pre-loop holds xi*q
  __shared__ u16 Ht[64][64];   // h0^T bf16 (B layout rows=e, cols=d)
  __shared__ int tsq[64], sgq[64], tsk[64], sgk[64];

  int t = threadIdx.x;
  int w = t >> 5, l = t & 31;
  int khalf = l >> 4, lr = l & 15;
  int qblk = blockIdx.x;
  int bh = blockIdx.y;
  int b = bh / H_, h = bh % H_;
  float logk = head_logk(h);
  int qrow0 = b * S_ + qblk * 64;

  // load Q (raw into Qs, xi-scaled into Ps)
  for (int i = t; i < 64 * 8; i += 128) {
    int r = i >> 3, cb = (i & 7) * 8;
    int grow = qrow0 + r;
    u16x8 qv = *(const u16x8*)&qb[(size_t)grow * E_ + h * 64 + cb];
    *(u16x8*)&Qs[r][cb] = qv;
    int sr = seg[grow];
    float xi = (sr == 0) ? expf((float)(ts[grow] + 1) * logk) : 0.f;
    u16x8 qx;
#pragma unroll
    for (int e = 0; e < 8; ++e) qx[e] = f2bf(xi * bf2f(qv[e]));
    *(u16x8*)&Ps[r][cb] = qx;
  }
  if (t < 64) { tsq[t] = ts[qrow0 + t]; sgq[t] = seg[qrow0 + t]; }
  const float* hp = hstate + ((((size_t)b * H_ + h) * L_ + layer) * 64) * 64;
  for (int i = t; i < 4096; i += 128) {
    int d = i >> 6, e = i & 63;
    Ht[e][d] = f2bf(hp[i]);
  }
  __syncthreads();

  v8f o[4];
#pragma unroll
  for (int et = 0; et < 4; ++et) o[et] = v8f_zero();

  // cross term: O = (xi * Q) @ h0   (A = Ps, B = Ht)
  {
    FragB a[2];
#pragma unroll
    for (int ks = 0; ks < 2; ++ks) {
      int r = w * 16 + lr;
      a[ks].h[0] = *(const u16x8*)&Ps[r][ks * 32 + khalf * 8];
      a[ks].h[1] = *(const u16x8*)&Ps[r][ks * 32 + 16 + khalf * 8];
    }
#pragma unroll
    for (int et = 0; et < 4; ++et) {
      int c = et * 16 + lr;
#pragma unroll
      for (int ks = 0; ks < 2; ++ks) {
        FragB bb;
        bb.h[0] = *(const u16x8*)&Ht[c][ks * 32 + khalf * 8];
        bb.h[1] = *(const u16x8*)&Ht[c][ks * 32 + 16 + khalf * 8];
        o[et] = __builtin_amdgcn_wmma_f32_16x16x32_bf16(
            false, a[ks].v, false, bb.v, (short)0, o[et], false, false);
      }
    }
  }

  // key blocks (timestep groups of 8 are 64-block aligned -> causal at block level)
  for (int mb = 0; mb <= qblk; ++mb) {
    __syncthreads();
    int mrow0 = b * S_ + mb * 64;
    // K tile: pure copy -> async to LDS (ASYNCcnt path)
    for (int i = t; i < 64 * 4; i += 128) {
      int r = i >> 2, cb = (i & 3) * 16;
      async_cp32(lds_off(&Ks[r][cb]),
                 &kb[(size_t)(mrow0 + r) * E_ + h * 64 + cb]);
    }
    for (int i = t; i < 4096; i += 128) {
      int m = i >> 6, e = i & 63;
      Vt[e][m] = vb[(size_t)(mrow0 + m) * E_ + h * 64 + e];
    }
    if (t < 64) { tsk[t] = ts[mrow0 + t]; sgk[t] = seg[mrow0 + t]; }
    asm volatile("s_wait_asynccnt 0x0" ::: "memory");
    __syncthreads();

    // S = Q K^T with decay mask -> Ps (rows of Ps are wave-private)
    FragB aq[2];
#pragma unroll
    for (int ks = 0; ks < 2; ++ks) {
      int r = w * 16 + lr;
      aq[ks].h[0] = *(const u16x8*)&Qs[r][ks * 32 + khalf * 8];
      aq[ks].h[1] = *(const u16x8*)&Qs[r][ks * 32 + 16 + khalf * 8];
    }
#pragma unroll
    for (int mt = 0; mt < 4; ++mt) {
      v8f s = v8f_zero();
      int c = mt * 16 + lr;
#pragma unroll
      for (int ks = 0; ks < 2; ++ks) {
        FragB bb;
        bb.h[0] = *(const u16x8*)&Ks[c][ks * 32 + khalf * 8];
        bb.h[1] = *(const u16x8*)&Ks[c][ks * 32 + 16 + khalf * 8];
        s = __builtin_amdgcn_wmma_f32_16x16x32_bf16(
            false, aq[ks].v, false, bb.v, (short)0, s, false, false);
      }
#pragma unroll
      for (int j = 0; j < 8; ++j) {
        int nl = w * 16 + j + khalf * 8;
        int dt = tsq[nl] - tsk[c];
        float sv = (dt >= 0 && sgq[nl] == sgk[c]) ? s[j] * expf((float)dt * logk) : 0.f;
        Ps[nl][c] = f2bf(sv);
      }
    }
    // O += P @ V
#pragma unroll
    for (int ks = 0; ks < 2; ++ks) {
      FragB ap;
      int r = w * 16 + lr;
      ap.h[0] = *(const u16x8*)&Ps[r][ks * 32 + khalf * 8];
      ap.h[1] = *(const u16x8*)&Ps[r][ks * 32 + 16 + khalf * 8];
#pragma unroll
      for (int et = 0; et < 4; ++et) {
        int c2 = et * 16 + lr;
        FragB bv;
        bv.h[0] = *(const u16x8*)&Vt[c2][ks * 32 + khalf * 8];
        bv.h[1] = *(const u16x8*)&Vt[c2][ks * 32 + 16 + khalf * 8];
        o[et] = __builtin_amdgcn_wmma_f32_16x16x32_bf16(
            false, ap.v, false, bv.v, (short)0, o[et], false, false);
      }
    }
  }

#pragma unroll
  for (int et = 0; et < 4; ++et)
#pragma unroll
    for (int j = 0; j < 8; ++j) {
      int nl = w * 16 + j + khalf * 8;
      int e = et * 16 + lr;
      ret[(size_t)(qrow0 + nl) * E_ + h * 64 + e] = o[et][j];
    }
}

// ---------------- recurrent state update per (b,h) ----------------
__global__ __launch_bounds__(128) void state_kernel(
    const u16* __restrict__ kb, const u16* __restrict__ vb,
    const int* __restrict__ ts, const int* __restrict__ seg,
    const float* __restrict__ hstate, int layer, float* __restrict__ newh) {
  __shared__ u16 Kt[64][32];  // A layout: rows=d, cols=m (eta-scaled k)
  __shared__ u16 Vt[64][32];  // B layout: rows=e, cols=m
  __shared__ int tsm[32], sgm[32];
  int t = threadIdx.x, w = t >> 5, l = t & 31;
  int khalf = l >> 4, lr = l & 15;
  int bh = blockIdx.x;
  int b = bh / H_, h = bh % H_;
  float logk = head_logk(h);
  int tmax = ts[b * S_ + S_ - 1];
  int segL = seg[b * S_ + S_ - 1];

  v8f acc[4];
#pragma unroll
  for (int et = 0; et < 4; ++et) acc[et] = v8f_zero();

  for (int mc = 0; mc < S_ / 32; ++mc) {
    __syncthreads();
    if (t < 32) { tsm[t] = ts[b * S_ + mc * 32 + t]; sgm[t] = seg[b * S_ + mc * 32 + t]; }
    __syncthreads();
    for (int i = t; i < 2048; i += 128) {
      int d = i >> 5, m = i & 31;
      int grow = b * S_ + mc * 32 + m;
      float eta = (sgm[m] == segL) ? expf((float)(tmax - tsm[m]) * logk) : 0.f;
      Kt[d][m] = f2bf(eta * bf2f(kb[(size_t)grow * E_ + h * 64 + d]));
      Vt[d][m] = vb[(size_t)grow * E_ + h * 64 + d];  // here d indexes e
    }
    __syncthreads();
    FragB a;
    int r = w * 16 + lr;
    a.h[0] = *(const u16x8*)&Kt[r][khalf * 8];
    a.h[1] = *(const u16x8*)&Kt[r][16 + khalf * 8];
#pragma unroll
    for (int et = 0; et < 4; ++et) {
      int c = et * 16 + lr;
      FragB bv;
      bv.h[0] = *(const u16x8*)&Vt[c][khalf * 8];
      bv.h[1] = *(const u16x8*)&Vt[c][16 + khalf * 8];
      acc[et] = __builtin_amdgcn_wmma_f32_16x16x32_bf16(
          false, a.v, false, bv.v, (short)0, acc[et], false, false);
    }
  }

  float keep = (segL == 0) ? 1.f : 0.f;
  float hdec = keep * expf((float)(tmax + 1) * logk);
  const float* hpp = hstate + ((((size_t)b * H_ + h) * L_ + layer) * 64) * 64;
  float* op = newh + ((((size_t)b * H_ + h) * L_ + layer) * 64) * 64;
#pragma unroll
  for (int et = 0; et < 4; ++et)
#pragma unroll
    for (int j = 0; j < 8; ++j) {
      int d = w * 16 + j + khalf * 8;
      int e = et * 16 + lr;
      op[d * 64 + e] = hpp[d * 64 + e] * hdec + acc[et][j];
    }
}

// ---------------- host orchestration ----------------
extern "C" void kernel_launch(void* const* d_in, const int* in_sizes, int n_in,
                              void* d_out, int out_size, void* d_ws, size_t ws_size,
                              hipStream_t stream) {
  (void)in_sizes; (void)n_in; (void)out_size; (void)ws_size;
  const float* obs    = (const float*)d_in[0];
  const float* hstate = (const float*)d_in[1];
  const int*   dones  = (const int*)d_in[2];
  const int*   tsid   = (const int*)d_in[3];
  const float* ln_enc = (const float*)d_in[4];
  const float* w_enc  = (const float*)d_in[5];
  const float* ln_sh  = (const float*)d_in[6];
  const float* wq     = (const float*)d_in[7];
  const float* wk     = (const float*)d_in[8];
  const float* wv     = (const float*)d_in[9];
  const float* wg     = (const float*)d_in[10];
  const float* wo     = (const float*)d_in[11];
  const float* gns    = (const float*)d_in[12];
  const float* gnb    = (const float*)d_in[13];
  const float* ln1    = (const float*)d_in[14];
  const float* ln2    = (const float*)d_in[15];
  const float* swg    = (const float*)d_in[16];
  const float* swl    = (const float*)d_in[17];
  const float* swo    = (const float*)d_in[18];
  const float* hw1    = (const float*)d_in[19];
  const float* hb1    = (const float*)d_in[20];
  const float* hln    = (const float*)d_in[21];
  const float* hw2    = (const float*)d_in[22];
  const float* hb2    = (const float*)d_in[23];

  float* vloc = (float*)d_out;
  float* xf   = vloc + NROWS;                 // [N,E] f32 (also final output x)
  float* newh = xf + (size_t)NROWS * E_;      // [B,H,L,64,64]

  char* wp = (char*)d_ws;
  auto alloc = [&](size_t bytes) -> void* {
    void* p = (void*)wp;
    wp += (bytes + 255) & ~(size_t)255;
    return p;
  };
  const size_t EE = (size_t)E_ * E_;
  u16* wq_t  = (u16*)alloc(L_ * EE * 2);
  u16* wk_t  = (u16*)alloc(L_ * EE * 2);
  u16* wv_t  = (u16*)alloc(L_ * EE * 2);
  u16* wg_t  = (u16*)alloc(L_ * EE * 2);
  u16* wo_t  = (u16*)alloc(L_ * EE * 2);
  u16* swg_t = (u16*)alloc(L_ * EE * 2);
  u16* swl_t = (u16*)alloc(L_ * EE * 2);
  u16* swo_t = (u16*)alloc(L_ * EE * 2);
  u16* wenc_t = (u16*)alloc((size_t)E_ * OBS_ * 2);
  u16* hw1_t  = (u16*)alloc(EE * 2);
  int* seg    = (int*)alloc((size_t)B_ * S_ * 4);
  u16* obsn   = (u16*)alloc((size_t)NROWS * OBS_ * 2);
  u16* xinb   = (u16*)alloc((size_t)NROWS * E_ * 2);
  float* xinf = (float*)alloc((size_t)NROWS * E_ * 4);
  u16* qbf    = (u16*)alloc((size_t)NROWS * E_ * 2);
  u16* kbf    = (u16*)alloc((size_t)NROWS * E_ * 2);
  u16* vbf    = (u16*)alloc((size_t)NROWS * E_ * 2);
  float* gf   = (float*)alloc((size_t)NROWS * E_ * 4);
  float* retf = (float*)alloc((size_t)NROWS * E_ * 4);
  u16* tb     = (u16*)alloc((size_t)NROWS * E_ * 2);
  float* rof  = (float*)alloc((size_t)NROWS * E_ * 4);
  float* x1f  = (float*)alloc((size_t)NROWS * E_ * 4);
  u16* x1b    = (u16*)alloc((size_t)NROWS * E_ * 2);
  float* gatef = (float*)alloc((size_t)NROWS * E_ * 4);
  float* linf  = (float*)alloc((size_t)NROWS * E_ * 4);
  float* ffnf  = (float*)alloc((size_t)NROWS * E_ * 4);
  float* zf    = (float*)alloc((size_t)NROWS * E_ * 4);

  auto launch_gemm = [&](const u16* A, const u16* Bt, int M, int N, int K,
                         float* oF, u16* oB, const float* bias, int act, float scale) {
    dim3 grid(N / 128, M / 128);
    gemm_bf16_kernel<<<grid, 256, 0, stream>>>(A, Bt, M, N, K, oF, oB, bias, act, scale);
  };

  seg_kernel<<<1, 32, 0, stream>>>(dones, seg);
  for (int i = 0; i < L_; ++i) {
    wconv_kernel<<<256, 256, 0, stream>>>(wq  + i * EE, wq_t  + i * EE, E_, E_);
    wconv_kernel<<<256, 256, 0, stream>>>(wk  + i * EE, wk_t  + i * EE, E_, E_);
    wconv_kernel<<<256, 256, 0, stream>>>(wv  + i * EE, wv_t  + i * EE, E_, E_);
    wconv_kernel<<<256, 256, 0, stream>>>(wg  + i * EE, wg_t  + i * EE, E_, E_);
    wconv_kernel<<<256, 256, 0, stream>>>(wo  + i * EE, wo_t  + i * EE, E_, E_);
    wconv_kernel<<<256, 256, 0, stream>>>(swg + i * EE, swg_t + i * EE, E_, E_);
    wconv_kernel<<<256, 256, 0, stream>>>(swl + i * EE, swl_t + i * EE, E_, E_);
    wconv_kernel<<<256, 256, 0, stream>>>(swo + i * EE, swo_t + i * EE, E_, E_);
  }
  wconv_kernel<<<64, 256, 0, stream>>>(w_enc, wenc_t, OBS_, E_);
  wconv_kernel<<<256, 256, 0, stream>>>(hw1, hw1_t, E_, E_);

  // encoder: x = gelu(rmsnorm(obs) @ w_enc)
  rmsnorm_kernel<<<NROWS, 256, 0, stream>>>(obs, nullptr, ln_enc, OBS_, nullptr, obsn);
  launch_gemm(obsn, wenc_t, NROWS, E_, OBS_, xf, nullptr, nullptr, 1, 1.f);

  for (int i = 0; i < L_; ++i) {
    rmsnorm_kernel<<<NROWS, 256, 0, stream>>>(xf, nullptr, ln_sh, E_, xinf, xinb);
    launch_gemm(xinb, wq_t + i * EE, NROWS, E_, E_, nullptr, qbf, nullptr, 0, 1.f);
    launch_gemm(xinb, wk_t + i * EE, NROWS, E_, E_, nullptr, kbf, nullptr, 0, 0.125f);
    launch_gemm(xinb, wv_t + i * EE, NROWS, E_, E_, nullptr, vbf, nullptr, 0, 1.f);
    launch_gemm(xinb, wg_t + i * EE, NROWS, E_, E_, gf, nullptr, nullptr, 0, 1.f);
    retention_kernel<<<dim3(S_ / 64, B_ * H_), 128, 0, stream>>>(
        qbf, kbf, vbf, tsid, seg, hstate, i, retf);
    state_kernel<<<B_ * H_, 128, 0, stream>>>(kbf, vbf, tsid, seg, hstate, i, newh);
    gn_silu_kernel<<<NROWS, 256, 0, stream>>>(retf, gf, gns + (size_t)i * E_,
                                              gnb + (size_t)i * E_, tb);
    launch_gemm(tb, wo_t + i * EE, NROWS, E_, E_, rof, nullptr, nullptr, 0, 1.f);
    rmsnorm_kernel<<<NROWS, 256, 0, stream>>>(xinf, rof, ln1 + (size_t)i * E_, E_, x1f, x1b);
    launch_gemm(x1b, swg_t + i * EE, NROWS, E_, E_, gatef, nullptr, nullptr, 0, 1.f);
    launch_gemm(x1b, swl_t + i * EE, NROWS, E_, E_, linf, nullptr, nullptr, 0, 1.f);
    silu_mul_kernel<<<512, 256, 0, stream>>>(gatef, linf, tb, NROWS * E_);
    launch_gemm(tb, swo_t + i * EE, NROWS, E_, E_, ffnf, nullptr, nullptr, 0, 1.f);
    rmsnorm_kernel<<<NROWS, 256, 0, stream>>>(x1f, ffnf, ln2 + (size_t)i * E_, E_, xf, xinb);
  }

  // head: z = gelu(x @ hw1 + hb1); v_loc = rmsnorm(z, hln) @ hw2 + hb2
  launch_gemm(xinb, hw1_t, NROWS, E_, E_, zf, nullptr, hb1, 1, 1.f);
  value_head_kernel<<<NROWS, 256, 0, stream>>>(zf, hln, hw2, hb2, vloc);
}
